// SpikeMultiheadAttention_35330400976901
// MI455X (gfx1250) — matrix operand order repaired and checked by reading, not compiled, not run on hardware
//
#include <hip/hip_runtime.h>

typedef __attribute__((ext_vector_type(16))) __bf16 v16bf;
typedef __attribute__((ext_vector_type(8)))  float  v8f;
typedef unsigned int u32x4 __attribute__((ext_vector_type(4)));
typedef int          i32x4 __attribute__((ext_vector_type(4)));
typedef int          i32x8 __attribute__((ext_vector_type(8)));

union Frag16 {
  uint4 q[2];
  v16bf v;
};

#define T_STEPS 8
#define S_LEN   512
#define B_SZ    8
#define D_DIM   512
#define H_NUM   8
#define HD      64
#define M_ROWS  4096                    // S*B
#define X_ELEMS (M_ROWS * D_DIM)        // 2,097,152
#define W_ELEMS (D_DIM * D_DIM)         // 262,144

#if __has_builtin(__builtin_amdgcn_tensor_load_to_lds)
#define HAVE_TDM 1
#else
#define HAVE_TDM 0
#endif

// ---------------------------------------------------------------------------
// Kernel 1: time-reduce sx over T (affine collapse of the scan), emit bf16.
// ---------------------------------------------------------------------------
__global__ __launch_bounds__(256) void k_sum_time(const float* __restrict__ sx,
                                                  __bf16* __restrict__ Xb) {
  int i = blockIdx.x * 256 + threadIdx.x;
  float a = 0.f;
#pragma unroll
  for (int t = 0; t < T_STEPS; ++t)
    a += sx[(size_t)t * X_ELEMS + i];
  Xb[i] = (__bf16)a;
}

__global__ __launch_bounds__(256) void k_cvt_bf16(const float* __restrict__ W,
                                                  __bf16* __restrict__ Wb) {
  int i = blockIdx.x * 256 + threadIdx.x;
  Wb[i] = (__bf16)W[i];
}

// ---------------------------------------------------------------------------
// Kernel 2: projection GEMM. C[m,n] = sum_d Xb[m,d]*Wb[n,d] + T*bias[n].
// One wave computes one 16x16 tile with K=512 (16 x v_wmma_f32_16x16x32_bf16).
// ---------------------------------------------------------------------------
__global__ __launch_bounds__(32) void k_proj(const __bf16* __restrict__ Xb,
                                             const __bf16* __restrict__ Wb,
                                             const float* __restrict__ bias,
                                             __bf16* __restrict__ Out) {
  const int tile = blockIdx.x;
  const int m0 = (tile >> 5) << 4;      // 256 m-tiles
  const int n0 = (tile & 31) << 4;      // 32 n-tiles
  const int lane = threadIdx.x & 31;
  const int lr = lane & 15;
  const int hi = lane >> 4;
  const int koffA = hi << 3;            // A frag: K{0-7,16-23} vs {8-15,24-31}
  const int koffB = hi << 3;            // B frag: K{0-7,16-23} vs {8-15,24-31}
  v8f c = {};
#pragma unroll 4
  for (int kk = 0; kk < D_DIM; kk += 32) {
    Frag16 a, bf;
    const __bf16* ap = Xb + (size_t)(m0 + lr) * D_DIM + kk + koffA;
    a.q[0] = *(const uint4*)(ap);
    a.q[1] = *(const uint4*)(ap + 16);
    const __bf16* bp = Wb + (size_t)(n0 + lr) * D_DIM + kk + koffB;
    bf.q[0] = *(const uint4*)(bp);
    bf.q[1] = *(const uint4*)(bp + 16);
    c = __builtin_amdgcn_wmma_f32_16x16x32_bf16(false, a.v, false, bf.v,
                                                (short)0, c, false, false);
  }
  const float bv = (float)T_STEPS * bias[n0 + lr];
#pragma unroll
  for (int v = 0; v < 8; ++v) {
    const int m = m0 + v + (hi << 3);
    Out[(size_t)m * D_DIM + n0 + lr] = (__bf16)(c[v] + bv);
  }
}

// ---------------------------------------------------------------------------
// Kernel 3: batched P = (scale/T) * Q K^T per head + fused epilogue.
// K panel (512 x 64 bf16 = 64 KB) is DMA'd into dynamic LDS by the Tensor
// Data Mover (one tensor_load_to_lds issued by wave 0, D# per ISA ch.8),
// overlapping the per-lane Q fragment loads. WMMA B operands then come from
// LDS. Epilogue (elementwise 32-unit MLP, row reductions, inv-MLP scaling)
// runs out of a 32 KB LDS P tile.
// ---------------------------------------------------------------------------
__device__ __forceinline__ float mlp32(float x, const float* w1, const float* b1,
                                       const float* w2, float b2) {
  float acc = b2;
#pragma unroll
  for (int n = 0; n < 32; ++n) {
    float h = fmaxf(fmaf(x, w1[n], b1[n]), 0.f);
    acc = fmaf(h, w2[n], acc);
  }
  return acc;
}

__device__ __forceinline__ float mlp64g(float x, const float* w1, const float* b1,
                                        const float* w2, const float* b2) {
  float acc = b2[0];
#pragma unroll 8
  for (int n = 0; n < 64; ++n) {
    float h = fmaxf(fmaf(x, w1[n], b1[n]), 0.f);
    acc = fmaf(h, w2[n], acc);
  }
  return acc;
}

__global__ __launch_bounds__(128) void k_attn(const __bf16* __restrict__ Qb,
                                              const __bf16* __restrict__ Kb,
                                              const float* __restrict__ ew1,
                                              const float* __restrict__ eb1,
                                              const float* __restrict__ ew2,
                                              const float* __restrict__ eb2,
                                              const float* __restrict__ iw1,
                                              const float* __restrict__ ib1,
                                              const float* __restrict__ iw2,
                                              const float* __restrict__ ib2,
                                              float* __restrict__ out) {
  __shared__ float P[16][S_LEN];        // 32 KB
  __shared__ float rs[16][8];
  __shared__ float fRow[16];
  __shared__ float sw1[32], sb1[32], sw2[32];
  __shared__ float sb2s[1];
  extern __shared__ char KdynRaw[];     // 64 KB dynamic: K panel [512][64] bf16
  __bf16* Kdyn = (__bf16*)KdynRaw;

  const int bh = blockIdx.x;            // 0..63  (b*H + h)
  const int st = blockIdx.y;            // 0..31  (s-tile)
  const int b  = bh >> 3;
  const int h  = bh & 7;
  const int tid  = threadIdx.x;
  const int wave = tid >> 5;
  const int lane = tid & 31;
  const int lr = lane & 15;
  const int hi = lane >> 4;

  if (tid < 32) { sw1[tid] = ew1[tid]; sb1[tid] = eb1[tid]; sw2[tid] = ew2[tid]; }
  if (tid == 0) sb2s[0] = eb2[0];

#if HAVE_TDM
  if (wave == 0) {
    // Tensor DMA descriptor (ISA ch.8): 2D tile, rows = K columns s' (512),
    // row length 64 bf16 contiguous, row stride B*D = 4096 elements.
    const unsigned long long ga =
        (unsigned long long)(uintptr_t)(Kb + (size_t)b * D_DIM + (size_t)h * HD);
    u32x4 g0;
    g0[0] = 1u;                                        // count=1, user mode
    g0[1] = __builtin_amdgcn_groupstaticsize();        // LDS addr = dynamic base
    g0[2] = (unsigned)(ga & 0xffffffffu);              // global_addr[31:0]
    g0[3] = (unsigned)((ga >> 32) & 0x01ffffffu)       // global_addr[56:32]
          | (2u << 30);                                // type = 2 ("image")
    i32x8 g1 = {0, 0, 0, 0, 0, 0, 0, 0};
    g1[0] = 0x10000;                                   // data_size = 1 (2 bytes)
    g1[1] = (int)((unsigned)HD << 16);                 // tensor_dim0 = 64
    g1[2] = (int)((unsigned)S_LEN << 16);              // tensor_dim1 = 512
    g1[3] = (int)((unsigned)HD << 16);                 // tile_dim0 = 64
    g1[4] = S_LEN;                                     // tile_dim1 = 512
    g1[5] = B_SZ * D_DIM;                              // tensor_dim0_stride = 4096
    i32x4 gz4 = {0, 0, 0, 0};                          // groups 2/3: 2D tensor
    i32x8 gz8 = {0, 0, 0, 0, 0, 0, 0, 0};              // clang-23 6-arg form pad
    __builtin_amdgcn_tensor_load_to_lds(g0, g1, gz4, gz4, gz8, 0);
    __builtin_amdgcn_s_wait_tensorcnt(0);
  }
#else
  // Fallback: cooperative copy global -> LDS (uint4 = 8 bf16 per transfer).
  for (int i = tid; i < S_LEN * HD / 8; i += 128) {
    const int c = i >> 3, j = (i & 7) << 3;
    *(uint4*)(Kdyn + (size_t)c * HD + j) =
        *(const uint4*)(Kb + ((size_t)c * B_SZ + b) * D_DIM + (size_t)h * HD + j);
  }
#endif

  // A operand: Q tile rows s0..s0+15, K = 64 -> two 16x32 fragments
  // (overlaps with the TDM transfer above).
  const int s0 = st << 4;
  const size_t qrow = ((size_t)(s0 + lr) * B_SZ + b) * D_DIM + (size_t)h * HD;
  Frag16 a0, a1;
  {
    const __bf16* ap = Qb + qrow + (hi << 3);
    a0.q[0] = *(const uint4*)(ap);       // K {0-7 | 8-15}
    a0.q[1] = *(const uint4*)(ap + 16);  // K {16-23 | 24-31}
    a1.q[0] = *(const uint4*)(ap + 32);  // K {32-39 | 40-47}
    a1.q[1] = *(const uint4*)(ap + 48);  // K {48-55 | 56-63}
  }
  const float sInv = 0.015625f;          // (1/sqrt(64)) / T
  __syncthreads();                       // K panel visible to all waves

#pragma unroll
  for (int tt = 0; tt < 8; ++tt) {
    const int c0 = (wave << 7) + (tt << 4);
    const __bf16* bp = Kdyn + (size_t)(c0 + lr) * HD + (hi << 3);
    Frag16 b0, b1;
    b0.q[0] = *(const uint4*)(bp);       // K {0-7,16-23 | 8-15,24-31}
    b0.q[1] = *(const uint4*)(bp + 16);
    b1.q[0] = *(const uint4*)(bp + 32);  // K {32-39,48-55 | 40-47,56-63}
    b1.q[1] = *(const uint4*)(bp + 48);
    v8f c = {};
    c = __builtin_amdgcn_wmma_f32_16x16x32_bf16(false, a0.v, false, b0.v,
                                                (short)0, c, false, false);
    c = __builtin_amdgcn_wmma_f32_16x16x32_bf16(false, a1.v, false, b1.v,
                                                (short)0, c, false, false);
#pragma unroll
    for (int v = 0; v < 8; ++v)
      P[v + (hi << 3)][c0 + lr] = c[v] * sInv;
  }
  __syncthreads();

  // Epilogue: thread -> (row = tid/8, 64-col chunk). Read tmax first, barrier,
  // then overwrite P in place with x_up.
  const int row = tid >> 3;
  const int cb  = (tid & 7) << 6;
  const float tmax = P[row][0];
  __syncthreads();

  float part = 0.f;
  const float b2v = sb2s[0];
  for (int j = 0; j < 64; ++j) {
    float p  = P[row][cb + j];
    float tp = fminf(p - (tmax - 1.0f), 3.0f);
    float xu = (tp > -20.0f) ? mlp32(tp, sw1, sb1, sw2, b2v) : 0.f;
    P[row][cb + j] = xu;
    part += xu;
  }
  rs[row][tid & 7] = part;
  __syncthreads();

  if (tid < 16) {
    float s = 0.f;
#pragma unroll
    for (int j = 0; j < 8; ++j) s += rs[tid][j];
    float pinv = mlp64g(s, iw1, ib1, iw2, ib2);
    float p2 = s * pinv;
    float f = pinv;
    if (p2 > 1.5f) f *= mlp64g(p2, iw1, ib1, iw2, ib2);
    fRow[tid] = f;
  }
  __syncthreads();

  const size_t obase = (size_t)bh * S_LEN * S_LEN + (size_t)s0 * S_LEN;
  for (int i = tid; i < 16 * S_LEN; i += 128) {
    const int r = i >> 9, cc = i & (S_LEN - 1);
    out[obase + (size_t)r * S_LEN + cc] = P[r][cc] * fRow[r];
  }
}

// ---------------------------------------------------------------------------
extern "C" void kernel_launch(void* const* d_in, const int* in_sizes, int n_in,
                              void* d_out, int out_size, void* d_ws, size_t ws_size,
                              hipStream_t stream) {
  (void)in_sizes; (void)n_in; (void)out_size; (void)ws_size;
  const float* sx  = (const float*)d_in[0];
  const float* Wq  = (const float*)d_in[1];
  const float* Wk  = (const float*)d_in[2];
  const float* bq  = (const float*)d_in[3];
  const float* bk  = (const float*)d_in[4];
  const float* ew1 = (const float*)d_in[5];
  const float* eb1 = (const float*)d_in[6];
  const float* ew2 = (const float*)d_in[7];
  const float* eb2 = (const float*)d_in[8];
  const float* iw1 = (const float*)d_in[9];
  const float* ib1 = (const float*)d_in[10];
  const float* iw2 = (const float*)d_in[11];
  const float* ib2 = (const float*)d_in[12];

  char* ws = (char*)d_ws;
  __bf16* Xb  = (__bf16*)(ws);                                             // 4 MB
  __bf16* Wqb = (__bf16*)(ws + (size_t)X_ELEMS * 2);                       // 0.5 MB
  __bf16* Wkb = (__bf16*)(ws + (size_t)X_ELEMS * 2 + (size_t)W_ELEMS * 2); // 0.5 MB
  __bf16* Qb  = (__bf16*)(ws + (size_t)X_ELEMS * 2 + (size_t)W_ELEMS * 4); // 4 MB
  __bf16* Kb  = (__bf16*)(ws + (size_t)X_ELEMS * 4 + (size_t)W_ELEMS * 4); // 4 MB

  k_sum_time<<<X_ELEMS / 256, 256, 0, stream>>>(sx, Xb);
  k_cvt_bf16<<<W_ELEMS / 256, 256, 0, stream>>>(Wq, Wqb);
  k_cvt_bf16<<<W_ELEMS / 256, 256, 0, stream>>>(Wk, Wkb);

  const int projTiles = (M_ROWS / 16) * (D_DIM / 16);  // 8192
  k_proj<<<projTiles, 32, 0, stream>>>(Xb, Wqb, bq, Qb);
  k_proj<<<projTiles, 32, 0, stream>>>(Xb, Wkb, bk, Kb);

  dim3 g(B_SZ * H_NUM, S_LEN / 16);                    // 64 x 32
  const size_t kdynBytes = (size_t)S_LEN * HD * 2;     // 64 KB dynamic LDS
  k_attn<<<g, 128, kdynBytes, stream>>>(Qb, Kb, ew1, eb1, ew2, eb2,
                                        iw1, ib1, iw2, ib2, (float*)d_out);
}
